// GINlayer_3564822855758
// MI455X (gfx1250) — matrix operand order repaired and checked by reading, not compiled, not run on hardware
//
#include <hip/hip_runtime.h>
#include <stdint.h>

#define N_NODES_C 50000
#define N_EDGES_C 800000
#define D_IN  128
#define D_HID 256
#define D_OUT 128

typedef __bf16 v16bf __attribute__((ext_vector_type(16)));
typedef __bf16 v4bf  __attribute__((ext_vector_type(4)));
typedef float  v8f   __attribute__((ext_vector_type(8)));
typedef unsigned int v4u __attribute__((ext_vector_type(4)));
typedef int    v8i   __attribute__((ext_vector_type(8)));
typedef int    v4i   __attribute__((ext_vector_type(4)));

union AFrag { v16bf v; v4u q[2]; };

// ---------------------------------------------------------------------------
// Kernel 1: convert f32 weights -> bf16, transposed to N-major so WMMA
// B-fragments are pairs of contiguous 16B LDS reads.
// W1 [128][256] -> T1 [256][128];  W2 [256][128] -> T2 [128][256]
// ---------------------------------------------------------------------------
__global__ void convert_weights(const float* __restrict__ W1n, const float* __restrict__ W2n,
                                const float* __restrict__ W1e, const float* __restrict__ W2e,
                                __bf16* __restrict__ T1n, __bf16* __restrict__ T2n,
                                __bf16* __restrict__ T1e, __bf16* __restrict__ T2e) {
    int i = blockIdx.x * blockDim.x + threadIdx.x;
    if (i >= D_IN * D_HID) return;
    { int k = i / D_HID, n = i % D_HID;             // W1 element [k][n]
      T1n[n * D_IN + k] = (__bf16)W1n[i];
      T1e[n * D_IN + k] = (__bf16)W1e[i]; }
    { int k = i / D_OUT, n = i % D_OUT;             // W2 element [k][n]
      T2n[n * D_HID + k] = (__bf16)W2n[i];
      T2e[n * D_HID + k] = (__bf16)W2e[i]; }
}

// ---------------------------------------------------------------------------
// Kernel 2: edge message scatter: nz[dst] += nh[src] * eh   (f32 atomics, L2)
// 32 lanes per edge, float4 per lane.
// ---------------------------------------------------------------------------
__global__ void scatter_msgs(const float* __restrict__ nh,
                             const float* __restrict__ eh,
                             const int*   __restrict__ ei,
                             float*       __restrict__ nz) {
    int gid = blockIdx.x * blockDim.x + threadIdx.x;
    int e = gid >> 5;
    if (e >= N_EDGES_C) return;
    int c = (gid & 31) * 4;
    int s = ei[e];
    int d = ei[N_EDGES_C + e];
    const float4 a = *(const float4*)(nh + (long long)s * D_IN + c);
    const float4 b = *(const float4*)(eh + (long long)e * D_IN + c);
    float* p = nz + (long long)d * D_IN + c;
    unsafeAtomicAdd(p + 0, a.x * b.x);
    unsafeAtomicAdd(p + 1, a.y * b.y);
    unsafeAtomicAdd(p + 2, a.z * b.z);
    unsafeAtomicAdd(p + 3, a.w * b.w);
}

// ---------------------------------------------------------------------------
// TDM: stage a 2D bf16 matrix [dim1 rows][dim0 cols] into LDS, inserting
// 4 DWORDs (8 bf16) of padding after every row for conflict-free ds_load_b128.
// pad_interval_code: 5 -> every 64 DWORDs (128-col rows), 6 -> every 128 DWORDs.
// pad_amount code 3 -> 4 DWORDs.
// ---------------------------------------------------------------------------
__device__ __forceinline__ void tdm_load_2d(unsigned lds_off, const void* gptr,
                                            unsigned dim0, unsigned dim1,
                                            unsigned pad_interval_code) {
    unsigned long long ga = (unsigned long long)(uintptr_t)gptr;
    v4u g0;
    g0.x = 1u;                                           // count=1, user descriptor
    g0.y = lds_off;                                      // lds_addr
    g0.z = (unsigned)(ga & 0xffffffffu);                 // global_addr[31:0]
    g0.w = (unsigned)((ga >> 32) & 0x01ffffffu) | (2u << 30);  // global_addr[56:32] | type=2
    v8i g1;
    g1[0] = (int)((1u << 16) | (1u << 20) | (pad_interval_code << 22) | (3u << 25));
    //              data_size=2B  pad_en       pad_interval              pad_amount=4DW
    g1[1] = (int)(dim0 << 16);          // tensor_dim0[15:0] @ bits 63:48
    g1[2] = (int)(dim1 << 16);          // tensor_dim1[15:0] @ bits 95:80
    g1[3] = (int)(dim0 << 16);          // tile_dim0 @ bits 127:112
    g1[4] = (int)(dim1 & 0xffffu);      // tile_dim1 @ bits 143:128 (tile_dim2=0 -> 2D)
    g1[5] = (int)dim0;                  // tensor_dim0_stride[31:0]
    g1[6] = 0;
    g1[7] = 0;
    v4i z4 = {0, 0, 0, 0};
    v8i z8 = {0, 0, 0, 0, 0, 0, 0, 0};
    __builtin_amdgcn_tensor_load_to_lds(g0, g1, z4, z4, z8, 0);
}

// ---------------------------------------------------------------------------
// Kernel 3: fused 2-layer MLP over a 128-row tile, bf16 WMMA, f32 accum.
//   mode 0 (nodes): x = (1+eps)*nh[r] + nz[r]
//   mode 1 (edges): x = (1+eps)*eh[r] + nz[src[r]] - nz[dst[r]]
//   out[r] = relu(x@W1 + b1) @ W2 + b2
// 256 threads = 8 wave32; wave w owns rows 16w..16w+15 of the tile.
// LDS: Xs[128][136] + Hs[128][264] + W1s[256][136] + W2s[128][264] bf16 = 234 KB.
// ---------------------------------------------------------------------------
__global__ __launch_bounds__(256, 1)
void mlp_kernel(int mode, int M,
                const float* __restrict__ X,
                const float* __restrict__ nz,
                const int*   __restrict__ ei,
                const float* __restrict__ epsp,
                const __bf16* __restrict__ W1T,   // [256][128] bf16 N-major
                const __bf16* __restrict__ W2T,   // [128][256] bf16 N-major
                const float* __restrict__ b1,
                const float* __restrict__ b2,
                float* __restrict__ out) {
    extern __shared__ char smem[];
    __bf16* Xs  = (__bf16*)(smem);            // 128*136*2 = 34816
    __bf16* Hs  = (__bf16*)(smem + 34816);    // 128*264*2 = 67584
    __bf16* W1s = (__bf16*)(smem + 102400);   // 256*136*2 = 69632
    __bf16* W2s = (__bf16*)(smem + 172032);   // 128*264*2 = 67584
    const int XS = 136, HS = 264, W1S = 136, W2S = 264;

    const int tid  = threadIdx.x;
    const int wv   = tid >> 5;
    const int lane = tid & 31;
    const int half = lane >> 4;
    const int lm   = lane & 15;
    const int rowbase = blockIdx.x * 128;

    // --- wave 0 kicks off TDM weight staging (async, TENSORcnt) ---
    if (wv == 0) {
        tdm_load_2d((unsigned)(uintptr_t)W1s, W1T, D_IN,  D_HID, 5u); // rows of 64 DW
        tdm_load_2d((unsigned)(uintptr_t)W2s, W2T, D_HID, D_OUT, 6u); // rows of 128 DW
    }

    // --- all 8 waves build the X tile meanwhile ---
    {
        const float scale = 1.0f + epsp[0];
        int i  = tid >> 1;
        int c0 = (tid & 1) * 64;
        int r  = rowbase + i;
        bool valid = (r < M);
        int s = 0, d = 0;
        if (valid && mode == 1) { s = ei[r]; d = ei[N_EDGES_C + r]; }
        #pragma unroll
        for (int j = 0; j < 64; j += 4) {
            float4 v = make_float4(0.f, 0.f, 0.f, 0.f);
            if (valid) {
                const float4 x4 = *(const float4*)(X + (long long)r * D_IN + c0 + j);
                if (mode == 0) {
                    const float4 z4 = *(const float4*)(nz + (long long)r * D_IN + c0 + j);
                    v.x = scale * x4.x + z4.x; v.y = scale * x4.y + z4.y;
                    v.z = scale * x4.z + z4.z; v.w = scale * x4.w + z4.w;
                } else {
                    const float4 zs = *(const float4*)(nz + (long long)s * D_IN + c0 + j);
                    const float4 zd = *(const float4*)(nz + (long long)d * D_IN + c0 + j);
                    v.x = scale * x4.x + zs.x - zd.x; v.y = scale * x4.y + zs.y - zd.y;
                    v.z = scale * x4.z + zs.z - zd.z; v.w = scale * x4.w + zs.w - zd.w;
                }
            }
            // native pack-convert (v_cvt_pk_bf16_f32) + single ds_store_b64
            *(v4bf*)(&Xs[i * XS + c0 + j]) =
                (v4bf){(__bf16)v.x, (__bf16)v.y, (__bf16)v.z, (__bf16)v.w};
        }
    }
    if (wv == 0) __builtin_amdgcn_s_wait_tensorcnt(0);
    __syncthreads();

    // --- GEMM1: H = relu(X @ W1 + b1) ; wave row-stripe, 16 N-tiles ---
    AFrag a1[4];
    {
        const __bf16* xb = Xs + (16 * wv + lm) * XS + 8 * half;
        #pragma unroll
        for (int k = 0; k < 4; ++k) {
            a1[k].q[0] = *(const v4u*)(xb + 32 * k);        // K = 32k + 8h .. +7
            a1[k].q[1] = *(const v4u*)(xb + 32 * k + 16);   // K = 32k + 16 + 8h .. +7
        }
    }
    #pragma unroll
    for (int n = 0; n < 16; ++n) {
        const int col = 16 * n + lm;
        const float bias = b1[col];
        v8f acc = {bias, bias, bias, bias, bias, bias, bias, bias};
        #pragma unroll
        for (int k = 0; k < 4; ++k) {
            AFrag bf;
            const __bf16* wb = W1s + col * W1S + 32 * k + 8 * half;
            bf.q[0] = *(const v4u*)(wb);
            bf.q[1] = *(const v4u*)(wb + 16);
            acc = __builtin_amdgcn_wmma_f32_16x16x32_bf16(
                false, a1[k].v, false, bf.v, (short)0, acc, false, false);
        }
        #pragma unroll
        for (int g = 0; g < 8; ++g) {
            Hs[(16 * wv + 8 * half + g) * HS + col] = (__bf16)fmaxf(acc[g], 0.0f);
        }
    }
    __syncthreads();

    // --- GEMM2: out = H @ W2 + b2 ; 8 N-tiles, K=256 ---
    AFrag a2[8];
    {
        const __bf16* hb = Hs + (16 * wv + lm) * HS + 8 * half;
        #pragma unroll
        for (int k = 0; k < 8; ++k) {
            a2[k].q[0] = *(const v4u*)(hb + 32 * k);
            a2[k].q[1] = *(const v4u*)(hb + 32 * k + 16);
        }
    }
    #pragma unroll
    for (int n = 0; n < 8; ++n) {
        const int col = 16 * n + lm;
        const float bias = b2[col];
        v8f acc = {bias, bias, bias, bias, bias, bias, bias, bias};
        #pragma unroll
        for (int k = 0; k < 8; ++k) {
            AFrag bf;
            const __bf16* wb = W2s + col * W2S + 32 * k + 8 * half;
            bf.q[0] = *(const v4u*)(wb);
            bf.q[1] = *(const v4u*)(wb + 16);
            acc = __builtin_amdgcn_wmma_f32_16x16x32_bf16(
                false, a2[k].v, false, bf.v, (short)0, acc, false, false);
        }
        #pragma unroll
        for (int g = 0; g < 8; ++g) {
            int r = rowbase + 16 * wv + 8 * half + g;
            if (r < M) out[(long long)r * D_OUT + col] = acc[g];
        }
    }
}

// ---------------------------------------------------------------------------
extern "C" void kernel_launch(void* const* d_in, const int* in_sizes, int n_in,
                              void* d_out, int out_size, void* d_ws, size_t ws_size,
                              hipStream_t stream) {
    const float* nh     = (const float*)d_in[0];
    const float* eh     = (const float*)d_in[1];
    const int*   ei     = (const int*)  d_in[2];
    const float* nf_W1  = (const float*)d_in[3];
    const float* nf_b1  = (const float*)d_in[4];
    const float* nf_W2  = (const float*)d_in[5];
    const float* nf_b2  = (const float*)d_in[6];
    const float* nf_eps = (const float*)d_in[7];
    const float* ef_W1  = (const float*)d_in[8];
    const float* ef_b1  = (const float*)d_in[9];
    const float* ef_W2  = (const float*)d_in[10];
    const float* ef_b2  = (const float*)d_in[11];
    const float* ef_eps = (const float*)d_in[12];
    float* out = (float*)d_out;

    // workspace layout
    char* ws = (char*)d_ws;
    float* nz = (float*)ws;                       // 50000*128*4 = 25,600,000 B
    __bf16* T1n = (__bf16*)(ws + 25600000);       // 32768 bf16 each
    __bf16* T2n = T1n + D_IN * D_HID;
    __bf16* T1e = T2n + D_IN * D_HID;
    __bf16* T2e = T1e + D_IN * D_HID;

    const size_t lds_bytes = 239616;
    (void)hipFuncSetAttribute(reinterpret_cast<const void*>(mlp_kernel),
                              hipFuncAttributeMaxDynamicSharedMemorySize, (int)lds_bytes);

    (void)hipMemsetAsync(nz, 0, (size_t)N_NODES_C * D_IN * sizeof(float), stream);

    convert_weights<<<(D_IN * D_HID + 255) / 256, 256, 0, stream>>>(
        nf_W1, nf_W2, ef_W1, ef_W2, T1n, T2n, T1e, T2e);

    scatter_msgs<<<(N_EDGES_C * 32) / 256, 256, 0, stream>>>(nh, eh, ei, nz);

    // node MLP -> out[0 : 50000*128]
    mlp_kernel<<<(N_NODES_C + 127) / 128, 256, lds_bytes, stream>>>(
        0, N_NODES_C, nh, nz, ei, nf_eps, T1n, T2n, nf_b1, nf_b2, out);

    // edge MLP -> out[50000*128 : ]
    mlp_kernel<<<N_EDGES_C / 128, 256, lds_bytes, stream>>>(
        1, N_EDGES_C, eh, nz, ei, ef_eps, T1e, T2e, ef_b1, ef_b2,
        out + (long long)N_NODES_C * D_OUT);
}
